// Attn_7009386627292
// MI455X (gfx1250) — matrix operand order repaired and checked by reading, not compiled, not run on hardware
//
#include <hip/hip_runtime.h>
#include <hip/hip_bf16.h>
#include <math.h>

typedef __attribute__((ext_vector_type(16))) _Float16     v16h;
typedef __attribute__((ext_vector_type(8)))  float        v8f;
typedef __attribute__((ext_vector_type(4)))  unsigned int v4u;
typedef __attribute__((ext_vector_type(8)))  unsigned int v8u;

constexpr int Bsz = 2, Sq = 2048, Dm = 1024, Hh = 16, Dk = 64;
constexpr int Mrows = Bsz * Sq;               // 4096

// ---------------------------------------------------------------- cast f32->f16
__global__ void cast4_f32_f16(const float* __restrict__ src,
                              _Float16* __restrict__ dst, int n) {
  int i = (blockIdx.x * blockDim.x + threadIdx.x) * 4;
  if (i + 3 < n) {
    float4 v = *(const float4*)(src + i);
    dst[i + 0] = (_Float16)v.x;
    dst[i + 1] = (_Float16)v.y;
    dst[i + 2] = (_Float16)v.z;
    dst[i + 3] = (_Float16)v.w;
  }
}

// ---------------------------------------------------------------- transpose+cast
// src[R][C] f32 (row-major)  ->  dst[C][R] f16 (row-major)
__global__ __launch_bounds__(256) void transpose_cast(
    const float* __restrict__ src, _Float16* __restrict__ dst, int R, int C) {
  __shared__ _Float16 tile[32][33];
  const int c0 = blockIdx.x * 32, r0 = blockIdx.y * 32;
  const int tx = threadIdx.x & 31, ty = threadIdx.x >> 5;  // 8 rows per pass
#pragma unroll
  for (int rr = ty; rr < 32; rr += 8)
    tile[rr][tx] = (_Float16)src[(size_t)(r0 + rr) * C + c0 + tx];
  __syncthreads();
#pragma unroll
  for (int rr = ty; rr < 32; rr += 8)
    dst[(size_t)(c0 + rr) * R + r0 + tx] = tile[tx][rr];
}

// ---------------------------------------------------------------- TDM helpers
// D# group0: count=1, lds_addr (bytes), global_addr (tile start), type=2.
__device__ __forceinline__ v4u tdm_g0(const _Float16* gsrc, unsigned lds_off) {
  const unsigned long long ga = (unsigned long long)(uintptr_t)gsrc;
  v4u g0;
  g0.x = 1u;
  g0.y = lds_off;
  g0.z = (unsigned)ga;
  g0.w = (unsigned)((ga >> 32) & 0x01FFFFFFu) | (2u << 30);
  return g0;
}
// D# group1: data_size=2B; tile_dim0=cols(halfs), tile_dim1=rows, 2D tile;
// tensor_dim0_stride=row stride (halfs); tensor dims huge so OOB never fires.
__device__ __forceinline__ v8u tdm_g1(unsigned rows, unsigned cols,
                                      unsigned stride_halfs) {
  v8u g1;
  g1.s0 = (1u << 16);
  g1.s1 = 0u;
  g1.s2 = (1u << 4);
  g1.s3 = (1u << 4) | (cols << 16);
  g1.s4 = rows;
  g1.s5 = stride_halfs;
  g1.s6 = 0u;
  g1.s7 = 0u;
  return g1;
}
__device__ __forceinline__ void tdm_issue(v4u g0, v8u g1) {
  const v4u gz = {0u, 0u, 0u, 0u};
  asm volatile("tensor_load_to_lds %0, %1, %2, %3"
               :: "s"(g0), "s"(g1), "s"(gz), "s"(gz) : "memory");
}

// ---------------------------------------------------------------- WMMA GEMM
// C[M,N] = A[M,K](f16, row-major) * W[K,N] given as Bt[N][K](f16) (+bias).
// Store modes: 0 = f16 [B,H,S,64] scaled; 1 = f32 [M,N]+bias; 2 = f16 [B,H,64,S].
// Block tile 128(M) x 64(N); 8 waves x (16x64); K-step 32.
// Tiles streamed by the Tensor Data Mover, double-buffered: every wave DMAs its
// own slice (A rows w*16..+16, B rows w*8..+8) and waits only its own TENSORcnt;
// no divergent gate around the EXEC-ignoring TDM op.
__global__ __launch_bounds__(256) void gemm_wmma(
    const _Float16* __restrict__ A, const _Float16* __restrict__ Bt,
    const float* __restrict__ bias, _Float16* __restrict__ outH,
    float* __restrict__ outF, int M, int N, int K, float scale, int mode) {
  __shared__ _Float16 As[2][128 * 32];   // 2 x 8 KB  (m-major: As[p][m][k])
  __shared__ _Float16 Bs[2][64 * 32];    // 2 x 4 KB  (n-major: Bs[p][n][k])
  const int tid  = threadIdx.x;
  const int wave = tid >> 5, lane = tid & 31;
  const int g = lane >> 4, ln = lane & 15;
  const int m0 = blockIdx.y * 128;
  const int n0 = blockIdx.x * 64;

  const _Float16* Ap = A  + (size_t)(m0 + wave * 16) * K;  // this wave's A slice
  const _Float16* Bp = Bt + (size_t)(n0 + wave * 8)  * K;  // this wave's B slice
  const unsigned aoff[2] = {(unsigned)(uintptr_t)&As[0][wave * 16 * 32],
                            (unsigned)(uintptr_t)&As[1][wave * 16 * 32]};
  const unsigned boff[2] = {(unsigned)(uintptr_t)&Bs[0][wave * 8 * 32],
                            (unsigned)(uintptr_t)&Bs[1][wave * 8 * 32]};
  const v8u g1a = tdm_g1(16u, 32u, (unsigned)K);
  const v8u g1b = tdm_g1(8u,  32u, (unsigned)K);

  v8f acc[4] = {};

  // prologue: stage first tile pair into buffer 0
  tdm_issue(tdm_g0(Ap, aoff[0]), g1a);
  tdm_issue(tdm_g0(Bp, boff[0]), g1b);

  int p = 0;
  for (int k0 = 0; k0 < K; k0 += 32, p ^= 1) {
    if (k0 + 32 < K) {
      // start DMA of the next tile pair into the other buffer, then wait only
      // for the current pair (in-order TENSORcnt: 4 outstanding -> wait <= 2)
      tdm_issue(tdm_g0(Ap + k0 + 32, aoff[p ^ 1]), g1a);
      tdm_issue(tdm_g0(Bp + k0 + 32, boff[p ^ 1]), g1b);
      __builtin_amdgcn_s_wait_tensorcnt(2);
    } else {
      __builtin_amdgcn_s_wait_tensorcnt(0);
    }
    __syncthreads();

    // A fragment: one contiguous 16-half run (k = 16*g + h)
    v16h af;
    const int arow = wave * 16 + ln;
#pragma unroll
    for (int h = 0; h < 16; ++h) {
      int kk = h + g * 16;
      af[h] = As[p][arow * 32 + kk];
    }
    // preload all four B fragments, then issue WMMAs back-to-back
    v16h bf[4];
#pragma unroll
    for (int j = 0; j < 4; ++j)
#pragma unroll
      for (int h = 0; h < 16; ++h)
        bf[j][h] = Bs[p][(j * 16 + ln) * 32 + h + 16 * g];
#pragma unroll
    for (int j = 0; j < 4; ++j)
      acc[j] = __builtin_amdgcn_wmma_f32_16x16x32_f16(
          false, af, false, bf[j], (short)0, acc[j], false, false);
    __syncthreads();
  }

  // store: D layout -> row m = i + 8*g, col n = j*16 + ln  (unsigned shift/mask)
  const unsigned mbase = (unsigned)(m0 + wave * 16 + 8 * g);
#pragma unroll
  for (int j = 0; j < 4; ++j) {
    const unsigned n = (unsigned)(n0 + j * 16 + ln);
    const float bval = bias[n];
    const unsigned hh = n >> 6, d = n & 63u;
#pragma unroll
    for (int i = 0; i < 8; ++i) {
      const unsigned m = mbase + i;
      const float val = (acc[j][i] + bval) * scale;
      const unsigned b = m >> 11, s = m & 2047u;   // Sq = 2048
      if (mode == 0) {
        outH[(((size_t)b * Hh + hh) * Sq + s) * Dk + d] = (_Float16)val;
      } else if (mode == 2) {
        outH[(((size_t)b * Hh + hh) * Dk + d) * Sq + s] = (_Float16)val;
      } else {
        outF[(size_t)m * N + n] = val;
      }
    }
  }
}

// ---------------------------------------------------------------- flash attention
// q,k: f16 [B,H,S,64] (q pre-scaled by 1/sqrt(64)); v: f16 [B,H,64,S] (transposed).
// O: f16 [B,S,H*64]. Block = 8 waves; wave w owns q rows [blk*128 + w*16, +16).
__global__ __launch_bounds__(256) void flash_attn(
    const _Float16* __restrict__ Q, const _Float16* __restrict__ Kt,
    const _Float16* __restrict__ Vt, _Float16* __restrict__ O) {
  __shared__ _Float16 Pl[8][16 * 32];  // per-wave P scratch (D-layout -> A-layout)
  const int tid  = threadIdx.x;
  const int wave = tid >> 5, lane = tid & 31;
  const int g = lane >> 4, ln = lane & 15;
  const int b = blockIdx.z, h = blockIdx.y;
  const int q0 = blockIdx.x * 128 + wave * 16;
  const size_t head  = ((size_t)b * Hh + h) * Sq;       // rows in [B,H,S,64]
  const size_t headv = ((size_t)b * Hh + h) * Dk * Sq;  // base in [B,H,64,S]

  // Q fragments: 16x64 = two 16x32 A-frags, kept in registers
  v16h qa[2];
#pragma unroll
  for (int t = 0; t < 2; ++t)
#pragma unroll
    for (int hh = 0; hh < 16; ++hh) {
      int kd = hh + g * 16 + t * 32;
      qa[t][hh] = Q[(head + q0 + ln) * 64 + kd];
    }

  v8f acc[4] = {};
  float mrow[8], lrow[8];
#pragma unroll
  for (int i = 0; i < 8; ++i) { mrow[i] = -1e30f; lrow[i] = 0.0f; }

  const int kmax = q0 + 15;  // causal limit for this wave's rows
  for (int k0 = 0; k0 <= kmax; k0 += 32) {
    // ---- scores: two 16x16 tiles; preload K frags, then 4 WMMAs
    v16h kb[2][2];
#pragma unroll
    for (int u = 0; u < 2; ++u) {
      int key  = k0 + u * 16 + ln;
      int keyc = key < Sq ? key : Sq - 1;
#pragma unroll
      for (int t = 0; t < 2; ++t)
#pragma unroll
        for (int hh = 0; hh < 16; ++hh)
          kb[u][t][hh] = Kt[(head + keyc) * 64 + hh + 16 * g + 32 * t];
    }
    v8f sc[2];
#pragma unroll
    for (int u = 0; u < 2; ++u) {
      v8f s = {};
      s = __builtin_amdgcn_wmma_f32_16x16x32_f16(
          false, qa[0], false, kb[u][0], (short)0, s, false, false);
      s = __builtin_amdgcn_wmma_f32_16x16x32_f16(
          false, qa[1], false, kb[u][1], (short)0, s, false, false);
      // causal mask (finite -inf: exp underflows to exactly 0)
      int key = k0 + u * 16 + ln;
#pragma unroll
      for (int i = 0; i < 8; ++i) {
        int row = q0 + i + 8 * g;
        if (key > row) s[i] = -1e30f;
      }
      sc[u] = s;
    }

    // ---- online softmax (row reductions across half-wave lanes)
    float alpha[8];
#pragma unroll
    for (int i = 0; i < 8; ++i) {
      float t = fmaxf(sc[0][i], sc[1][i]);
#pragma unroll
      for (int msk = 1; msk < 16; msk <<= 1) t = fmaxf(t, __shfl_xor(t, msk, 32));
      float nm = fmaxf(mrow[i], t);
      alpha[i] = __expf(mrow[i] - nm);
      mrow[i] = nm;
      float p0 = __expf(sc[0][i] - nm);
      float p1 = __expf(sc[1][i] - nm);
      sc[0][i] = p0; sc[1][i] = p1;
      float rs = p0 + p1;
#pragma unroll
      for (int msk = 1; msk < 16; msk <<= 1) rs += __shfl_xor(rs, msk, 32);
      lrow[i] = lrow[i] * alpha[i] + rs;
    }
#pragma unroll
    for (int j = 0; j < 4; ++j)
#pragma unroll
      for (int i = 0; i < 8; ++i) acc[j][i] *= alpha[i];

    // ---- P: D-layout -> LDS (row-major 16x32) -> A-fragment
    asm volatile("" ::: "memory");
#pragma unroll
    for (int u = 0; u < 2; ++u)
#pragma unroll
      for (int i = 0; i < 8; ++i)
        Pl[wave][(i + 8 * g) * 32 + u * 16 + ln] = (_Float16)sc[u][i];
    asm volatile("s_wait_dscnt 0" ::: "memory");  // LDS in-order per wave
    v16h pa;
#pragma unroll
    for (int hh = 0; hh < 16; ++hh) {
      int kk = hh + g * 16;
      pa[hh] = Pl[wave][ln * 32 + kk];
    }
    asm volatile("" ::: "memory");

    // ---- O += P(16x32) * V(32x64): preload 4 V frags (keys contiguous), 4 WMMAs
    v16h vb[4];
#pragma unroll
    for (int j = 0; j < 4; ++j)
#pragma unroll
      for (int hh = 0; hh < 16; ++hh)
        vb[j][hh] = Vt[headv + (size_t)(j * 16 + ln) * Sq + k0 + hh + 16 * g];
#pragma unroll
    for (int j = 0; j < 4; ++j)
      acc[j] = __builtin_amdgcn_wmma_f32_16x16x32_f16(
          false, pa, false, vb[j], (short)0, acc[j], false, false);

    if (k0 + 32 <= kmax) {  // hint next tiles toward the caches
      __builtin_prefetch(&Kt[(head + k0 + 32) * 64], 0, 0);
      __builtin_prefetch(&Vt[headv + (size_t)ln * Sq + k0 + 32], 0, 0);
    }
  }

  // ---- epilogue: O[b, s, h*64 + n] = acc / l
#pragma unroll
  for (int j = 0; j < 4; ++j)
#pragma unroll
    for (int i = 0; i < 8; ++i) {
      int s   = q0 + i + 8 * g;
      int col = h * 64 + j * 16 + ln;
      O[((size_t)b * Sq + s) * (Hh * Dk) + col] = (_Float16)(acc[j][i] / lrow[i]);
    }
}

// ---------------------------------------------------------------- launch
extern "C" void kernel_launch(void* const* d_in, const int* in_sizes, int n_in,
                              void* d_out, int out_size, void* d_ws, size_t ws_size,
                              hipStream_t stream) {
  (void)in_sizes; (void)n_in; (void)out_size; (void)ws_size;
  const float* x  = (const float*)d_in[0];
  const float* Wq = (const float*)d_in[1];
  const float* bq = (const float*)d_in[2];
  const float* Wk = (const float*)d_in[3];
  const float* bk = (const float*)d_in[4];
  const float* Wv = (const float*)d_in[5];
  const float* bv = (const float*)d_in[6];
  const float* Wo = (const float*)d_in[7];
  const float* bo = (const float*)d_in[8];
  float* out = (float*)d_out;

  const size_t XN = (size_t)Mrows * Dm;        // 4096*1024
  const size_t WN = (size_t)Dm * Dm;           // 1024*1024
  const size_t QN = (size_t)Bsz * Hh * Sq * Dk;

  _Float16* xh  = (_Float16*)d_ws;
  _Float16* wqt = xh  + XN;   // transposed f16 weights [N][K]
  _Float16* wkt = wqt + WN;
  _Float16* wvt = wkt + WN;
  _Float16* wot = wvt + WN;
  _Float16* qh  = wot + WN;
  _Float16* kh  = qh  + QN;
  _Float16* vt  = kh  + QN;   // [B,H,64,S]
  _Float16* oh  = vt  + QN;   // also serves as finite pad for V tail overrun

  cast4_f32_f16<<<(int)(XN / 1024), 256, 0, stream>>>(x, xh, (int)XN);
  dim3 tgrid(Dm / 32, Dm / 32);
  transpose_cast<<<tgrid, 256, 0, stream>>>(Wq, wqt, Dm, Dm);
  transpose_cast<<<tgrid, 256, 0, stream>>>(Wk, wkt, Dm, Dm);
  transpose_cast<<<tgrid, 256, 0, stream>>>(Wv, wvt, Dm, Dm);
  transpose_cast<<<tgrid, 256, 0, stream>>>(Wo, wot, Dm, Dm);

  dim3 ggrid(Dm / 64, Mrows / 128);  // (16, 32)
  const float qscale = 0.125f;       // 1/sqrt(64), applied after bias (matches ref)
  gemm_wmma<<<ggrid, 256, 0, stream>>>(xh, wqt, bq, qh, nullptr, Mrows, Dm, Dm, qscale, 0);
  gemm_wmma<<<ggrid, 256, 0, stream>>>(xh, wkt, bk, kh, nullptr, Mrows, Dm, Dm, 1.0f, 0);
  gemm_wmma<<<ggrid, 256, 0, stream>>>(xh, wvt, bv, vt, nullptr, Mrows, Dm, Dm, 1.0f, 2);

  flash_attn<<<dim3(Sq / 128, Hh, Bsz), 256, 0, stream>>>(qh, kh, vt, oh);

  gemm_wmma<<<ggrid, 256, 0, stream>>>(oh, wot, bo, nullptr, out, Mrows, Dm, Dm, 1.0f, 1);
}
